// Attention_22290880266753
// MI455X (gfx1250) — matrix operand order repaired and checked by reading, not compiled, not run on hardware
//
#include <hip/hip_runtime.h>

#define B_ 2
#define N_ 2048
#define C_ 1024
#define H_ 16
#define D_ 64

typedef __attribute__((ext_vector_type(16))) __bf16 v16bf;
typedef __attribute__((ext_vector_type(8)))  float  v8f;
typedef __attribute__((ext_vector_type(4)))  unsigned v4u_;
typedef __attribute__((ext_vector_type(8)))  int      v8i_;
typedef __attribute__((ext_vector_type(4)))  int      v4i_;

union AFrag { v16bf v; unsigned u[8]; unsigned short h[16]; };
union Acc   { v8f v; float f[8]; };

__device__ __forceinline__ unsigned short f2bf(float f) {
  union { float f; unsigned u; } x; x.f = f;
  unsigned r = x.u + 0x7fffu + ((x.u >> 16) & 1u);   // round-to-nearest-even
  return (unsigned short)(r >> 16);
}

// A-matrix (16-bit, 16x32) per-lane K base for u32 slot v (ISA 7.12.2 table)
__device__ __forceinline__ int kbaseA(int lane, int v) {
  return ((v & 3) * 2) + ((v >> 2) * 16) + ((lane >> 4) * 8);
}

static __device__ __forceinline__ v8f wmma_bf16(const AFrag& a, const AFrag& b, v8f c) {
  return __builtin_amdgcn_wmma_f32_16x16x32_bf16(false, a.v, false, b.v, (short)0, c, false, false);
}

// ---------------------------------------------------- TDM 2-D tile -> LDS
// tile_d1 x tile_d0 tile of 2-byte elements; source row pitch stride0 elems.
// pad16: insert 4 DWORDs of LDS padding after every 16 DWORDs (64B rows ->
// 80B LDS row pitch, i.e. [..][40] ushort arrays).
__device__ __forceinline__ void tdm_load_2d(const unsigned short* gsrc, unsigned lds_off,
                                            unsigned tensor_d0, unsigned tensor_d1,
                                            unsigned tile_d0, unsigned tile_d1,
                                            unsigned stride0, bool pad16)
{
  unsigned long long ga = (unsigned long long)(size_t)gsrc;
  v4u_ g0;
  g0.x = 1u;                                    // count=1, user descriptor
  g0.y = lds_off;                               // lds_addr [63:32]
  g0.z = (unsigned)ga;                          // global_addr [95:64]
  g0.w = (unsigned)(ga >> 32) | (2u << 30);     // global_addr hi + type=2
  unsigned w0 = (1u << 16);                     // data_size = 2 bytes
  if (pad16) w0 |= (1u << 20) | (3u << 22) | (3u << 25); // pad_en, every 16 DW, +4 DW
  v8i_ g1;
  g1[0] = (int)w0;
  g1[1] = (int)(tensor_d0 << 16);               // tensor_dim0 lo16 @ bits 63:48
  g1[2] = (int)((tensor_d0 >> 16) | (tensor_d1 << 16));   // d0 hi | d1 lo
  g1[3] = (int)((tensor_d1 >> 16) | (tile_d0 << 16));     // d1 hi | tile_dim0
  g1[4] = (int)(tile_d1 & 0xffffu);             // tile_dim1 (tile_dim2 = 0)
  g1[5] = (int)stride0;                         // tensor_dim0_stride lo32
  g1[6] = 0;                                    // stride hi / dim1_stride
  g1[7] = 0;
  v4i_ z4 = {0, 0, 0, 0};
#if __clang_major__ >= 23
  v8i_ z8 = {0, 0, 0, 0, 0, 0, 0, 0};
  __builtin_amdgcn_tensor_load_to_lds(g0, g1, z4, z4, z8, 0);
#else
  __builtin_amdgcn_tensor_load_to_lds(g0, g1, z4, z4, 0);
#endif
}

// ---------------------------------------------------------------- convert
__global__ void cvt_kernel(const float* __restrict__ src, unsigned short* __restrict__ dst, int n) {
  int i = (blockIdx.x * blockDim.x + threadIdx.x) * 4;
  if (i + 3 < n) {
    float4 v = *(const float4*)(src + i);
    dst[i + 0] = f2bf(v.x); dst[i + 1] = f2bf(v.y);
    dst[i + 2] = f2bf(v.z); dst[i + 3] = f2bf(v.w);
  }
}

// w [K,Nn] f32 -> wT [Nn,K] bf16 (32x32 LDS tile transpose, 256 threads)
__global__ __launch_bounds__(256) void cvt_transpose_kernel(
    const float* __restrict__ src, unsigned short* __restrict__ dst, int K, int Nn)
{
  __shared__ unsigned short tile[32][33];
  const int kt = blockIdx.x * 32, nt = blockIdx.y * 32;
  const int tx = threadIdx.x & 31, ty = threadIdx.x >> 5;
  #pragma unroll
  for (int i = 0; i < 4; ++i) {
    const int k = ty + i * 8;
    tile[k][tx] = f2bf(src[(size_t)(kt + k) * Nn + nt + tx]);
  }
  __syncthreads();
  #pragma unroll
  for (int i = 0; i < 4; ++i) {
    const int n = ty + i * 8;
    dst[(size_t)(nt + n) * K + kt + tx] = tile[tx][n];
  }
}

// ------------------------------------------------- shared WMMA GEMM mainloop
// Block tile: M=128 (8 waves x 16 rows), N=64, K-step 32.
// A row-major [M,K]; BT row-major [N,K] (pre-transposed weights).
// Both tiles DMA'd into double-buffered padded LDS by the TDM.
__device__ __forceinline__ void gemm_mainloop(
    const unsigned short* __restrict__ A, const unsigned short* __restrict__ BT,
    int K, int mbase, int nbase,
    unsigned short (*As)[128][40], unsigned short (*BsT)[64][40], Acc acc[4])
{
  const int tid = threadIdx.x;
  const int lane = tid & 31;
  const int wave = tid >> 5;
  for (int j = 0; j < 4; ++j)
    for (int i = 0; i < 8; ++i) acc[j].f[i] = 0.0f;

  const unsigned short* aroot = A + (size_t)mbase * K;
  const unsigned short* broot = BT + (size_t)nbase * K;

  if (wave == 0) {   // prefetch k-chunk 0 into buffer 0
    tdm_load_2d(aroot, (unsigned)(size_t)&As[0][0][0],
                (unsigned)K, 1u << 20, 32u, 128u, (unsigned)K, true);
    tdm_load_2d(broot, (unsigned)(size_t)&BsT[0][0][0],
                (unsigned)K, 1u << 20, 32u, 64u, (unsigned)K, true);
  }

  const int NK = K / 32;
  for (int kc = 0; kc < NK; ++kc) {
    const int cur = kc & 1;
    if (wave == 0) __builtin_amdgcn_s_wait_tensorcnt(0);  // chunk kc landed
    __syncthreads();                                      // publish LDS
    if (wave == 0 && (kc + 1) < NK) {                     // prefetch kc+1
      const int nb = cur ^ 1;
      const int k0 = (kc + 1) * 32;
      tdm_load_2d(aroot + k0, (unsigned)(size_t)&As[nb][0][0],
                  (unsigned)K, 1u << 20, 32u, 128u, (unsigned)K, true);
      tdm_load_2d(broot + k0, (unsigned)(size_t)&BsT[nb][0][0],
                  (unsigned)K, 1u << 20, 32u, 64u, (unsigned)K, true);
    }

    AFrag af;
    const int mrow = wave * 16 + (lane & 15);
    #pragma unroll
    for (int v = 0; v < 8; ++v)
      af.u[v] = *(const unsigned*)&As[cur][mrow][kbaseA(lane, v)];
    #pragma unroll
    for (int j = 0; j < 4; ++j) {
      AFrag bf;
      const int ncol = j * 16 + (lane & 15);
      const int koff = (lane >> 4) * 16;
      #pragma unroll
      for (int v = 0; v < 8; ++v)
        bf.u[v] = *(const unsigned*)&BsT[cur][ncol][kbaseA(lane, v)];
      acc[j].v = wmma_bf16(af, bf, acc[j].v);
    }
    __syncthreads();   // all waves done with buffer `cur` before refill
  }
}

// -------------------------------------- QKV GEMM + fused per-head RMSNorm
// grid = (32, 48) where blockIdx.y = s*16 + h
__global__ __launch_bounds__(256) void qkv_gemm_kernel(
    const unsigned short* __restrict__ xbf, const unsigned short* __restrict__ wTbf,
    const float* __restrict__ b_qkv, const float* __restrict__ q_w, const float* __restrict__ k_w,
    unsigned short* __restrict__ qbf, unsigned short* __restrict__ kbf,
    unsigned short* __restrict__ vtbf)
{
  __shared__ unsigned short As[2][128][40];
  __shared__ unsigned short BsT[2][64][40];
  Acc acc[4];
  const int mbase = blockIdx.x * 128;
  const int s = blockIdx.y >> 4;
  const int h = blockIdx.y & 15;
  const int nbase = blockIdx.y * 64;
  gemm_mainloop(xbf, wTbf, C_, mbase, nbase, As, BsT, acc);

  const int tid = threadIdx.x, lane = tid & 31, wave = tid >> 5;
  const int nc = lane & 15;
  float bias[4];
  #pragma unroll
  for (int j = 0; j < 4; ++j) bias[j] = b_qkv[nbase + j * 16 + nc];
  #pragma unroll
  for (int j = 0; j < 4; ++j)
    #pragma unroll
    for (int r = 0; r < 8; ++r) acc[j].f[r] += bias[j];

  if (s < 2) {
    const float* w = (s == 0) ? q_w : k_w;
    float wv[4];
    #pragma unroll
    for (int j = 0; j < 4; ++j) wv[j] = w[j * 16 + nc];
    unsigned short* dst = (s == 0) ? qbf : kbf;
    #pragma unroll
    for (int r = 0; r < 8; ++r) {
      float ss = 0.f;
      #pragma unroll
      for (int j = 0; j < 4; ++j) ss += acc[j].f[r] * acc[j].f[r];
      #pragma unroll
      for (int msk = 1; msk < 16; msk <<= 1) ss += __shfl_xor(ss, msk, 32);
      const float scale = rsqrtf(ss * (1.0f / 64.0f) + 1e-6f);
      const int m = mbase + wave * 16 + r + 8 * (lane >> 4);
      const int b = m >> 11, n = m & (N_ - 1);
      const int bh = b * H_ + h;
      unsigned short* drow = dst + (((size_t)bh * N_ + n) << 6);
      #pragma unroll
      for (int j = 0; j < 4; ++j)
        drow[j * 16 + nc] = f2bf(acc[j].f[r] * scale * wv[j]);
    }
  } else { // v: store transposed [BH, D, N]
    #pragma unroll
    for (int r = 0; r < 8; ++r) {
      const int m = mbase + wave * 16 + r + 8 * (lane >> 4);
      const int b = m >> 11, n = m & (N_ - 1);
      const int bh = b * H_ + h;
      #pragma unroll
      for (int j = 0; j < 4; ++j) {
        const int d = j * 16 + nc;
        vtbf[(((size_t)bh * D_ + d) << 11) + n] = f2bf(acc[j].f[r]);
      }
    }
  }
}

// ------------------------------------------------------- flash attention
// grid = (16, 32); TDM double-buffers K/V^T chunks of 32 kv into LDS.
__global__ __launch_bounds__(256) void flash_kernel(
    const unsigned short* __restrict__ qbf, const unsigned short* __restrict__ kbf,
    const unsigned short* __restrict__ vtbf, unsigned short* __restrict__ attnbf)
{
  __shared__ unsigned short Ks[2][32][64];    // K chunk [kv][d]
  __shared__ unsigned short VTs[2][64][32];   // V^T chunk [d][kv]
  __shared__ unsigned short Ps[8][16][36];    // per-wave P relayout scratch
  const int tid = threadIdx.x, lane = tid & 31, wave = tid >> 5;
  const int bh = blockIdx.y;
  const int b = bh >> 4, h = bh & 15;
  const int qbase = blockIdx.x * 128 + wave * 16;
  const int nc = lane & 15;

  AFrag qf[2];   // Q 16x64 in two K=32 A-fragments
  {
    const unsigned short* qrow = qbf + (((size_t)bh * N_ + qbase + nc) << 6);
    #pragma unroll
    for (int t = 0; t < 2; ++t)
      #pragma unroll
      for (int v = 0; v < 8; ++v)
        qf[t].u[v] = *(const unsigned*)(qrow + kbaseA(lane, v) + t * 32);
  }
  Acc o[4];
  for (int j = 0; j < 4; ++j) for (int i = 0; i < 8; ++i) o[j].f[i] = 0.f;
  float mrun[8], lrun[8];
  for (int r = 0; r < 8; ++r) { mrun[r] = -1e30f; lrun[r] = 0.f; }

  const unsigned short* kroot = kbf + (((size_t)bh * N_) << 6);   // [N,64]
  const unsigned short* vroot = vtbf + (((size_t)bh * D_) << 11); // [64,N]

  if (wave == 0) {   // prefetch chunk 0 into buffer 0 via TDM
    tdm_load_2d(kroot, (unsigned)(size_t)&Ks[0][0][0], 64u, (unsigned)N_, 64u, 32u, 64u, false);
    tdm_load_2d(vroot, (unsigned)(size_t)&VTs[0][0][0], (unsigned)N_, 64u, 32u, 64u, (unsigned)N_, false);
  }

  const int NCHUNK = N_ / 32;
  for (int ci = 0; ci < NCHUNK; ++ci) {
    const int cur = ci & 1;
    if (wave == 0) __builtin_amdgcn_s_wait_tensorcnt(0);  // chunk ci landed
    __syncthreads();                                      // publish LDS
    if (wave == 0 && (ci + 1) < NCHUNK) {                 // prefetch ci+1
      const int nb = cur ^ 1;
      const int kvn = (ci + 1) * 32;
      tdm_load_2d(kroot + ((size_t)kvn << 6), (unsigned)(size_t)&Ks[nb][0][0],
                  64u, (unsigned)N_, 64u, 32u, 64u, false);
      tdm_load_2d(vroot + kvn, (unsigned)(size_t)&VTs[nb][0][0],
                  (unsigned)N_, 64u, 32u, 64u, (unsigned)N_, false);
    }

    Acc sA[2];
    #pragma unroll
    for (int kvt = 0; kvt < 2; ++kvt) {       // two 16-kv S tiles
      for (int i = 0; i < 8; ++i) sA[kvt].f[i] = 0.f;
      const int nkv = kvt * 16 + nc;
      #pragma unroll
      for (int t = 0; t < 2; ++t) {           // d = 64 -> two K=32 steps
        AFrag kf;
        #pragma unroll
        for (int v = 0; v < 8; ++v) {
          const int kd = kbaseA(lane, v) + t * 32;
          kf.u[v] = *(const unsigned*)&Ks[cur][nkv][kd];
        }
        sA[kvt].v = wmma_bf16(qf[t], kf, sA[kvt].v);
      }
      #pragma unroll
      for (int i = 0; i < 8; ++i) sA[kvt].f[i] *= 0.125f;  // d^-0.5
    }

    // online softmax (row reductions live in a 16-lane half)
    #pragma unroll
    for (int r = 0; r < 8; ++r) {
      float rm = fmaxf(sA[0].f[r], sA[1].f[r]);
      #pragma unroll
      for (int msk = 1; msk < 16; msk <<= 1) rm = fmaxf(rm, __shfl_xor(rm, msk, 32));
      const float nm = fmaxf(mrun[r], rm);
      const float alpha = __expf(mrun[r] - nm);
      mrun[r] = nm;
      lrun[r] *= alpha;
      #pragma unroll
      for (int j = 0; j < 4; ++j) o[j].f[r] *= alpha;
    }

    #pragma unroll
    for (int r = 0; r < 8; ++r) {
      const int prow = r + 8 * (lane >> 4);
      float rs = 0.f;
      #pragma unroll
      for (int kvt = 0; kvt < 2; ++kvt) {
        const float p = __expf(sA[kvt].f[r] - mrun[r]);
        rs += p;
        Ps[wave][prow][kvt * 16 + nc] = f2bf(p);    // C-layout -> LDS
      }
      #pragma unroll
      for (int msk = 1; msk < 16; msk <<= 1) rs += __shfl_xor(rs, msk, 32);
      lrun[r] += rs;
    }

    AFrag pf;   // re-read P in A-layout (same-wave LDS is in-order)
    #pragma unroll
    for (int v = 0; v < 8; ++v)
      pf.u[v] = *(const unsigned*)&Ps[wave][nc][kbaseA(lane, v)];

    #pragma unroll
    for (int j = 0; j < 4; ++j) {             // O += P (16x32) * V (32x16*4)
      AFrag vf;
      #pragma unroll
      for (int v = 0; v < 8; ++v) {
        const int kk = kbaseA(lane, v);
        vf.u[v] = *(const unsigned*)&VTs[cur][j * 16 + nc][kk];
      }
      o[j].v = wmma_bf16(pf, vf, o[j].v);
    }
    __syncthreads();   // all waves done with buffer `cur` before it is refilled
  }

  #pragma unroll
  for (int r = 0; r < 8; ++r) {
    const float inv = 1.0f / lrun[r];
    const int n = qbase + r + 8 * (lane >> 4);
    unsigned short* orow = attnbf + (((size_t)b * N_ + n) << 10) + h * D_;
    #pragma unroll
    for (int j = 0; j < 4; ++j)
      orow[j * 16 + nc] = f2bf(o[j].f[r] * inv);
  }
}

// -------------------------------------------------------------- projection
__global__ __launch_bounds__(256) void proj_gemm_kernel(
    const unsigned short* __restrict__ abf, const unsigned short* __restrict__ wTbf,
    const float* __restrict__ bias, float* __restrict__ out)
{
  __shared__ unsigned short As[2][128][40];
  __shared__ unsigned short BsT[2][64][40];
  Acc acc[4];
  const int mbase = blockIdx.x * 128;
  const int nbase = blockIdx.y * 64;
  gemm_mainloop(abf, wTbf, C_, mbase, nbase, As, BsT, acc);

  const int tid = threadIdx.x, lane = tid & 31, wave = tid >> 5;
  const int nc = lane & 15;
  #pragma unroll
  for (int r = 0; r < 8; ++r) {
    const int m = mbase + wave * 16 + r + 8 * (lane >> 4);
    float* orow = out + (size_t)m * C_ + nbase;
    #pragma unroll
    for (int j = 0; j < 4; ++j)
      orow[j * 16 + nc] = acc[j].f[r] + bias[nbase + j * 16 + nc];
  }
}

// ------------------------------------------------------------------ launch
extern "C" void kernel_launch(void* const* d_in, const int* in_sizes, int n_in,
                              void* d_out, int out_size, void* d_ws, size_t ws_size,
                              hipStream_t stream)
{
  const float* x      = (const float*)d_in[0];
  const float* w_qkv  = (const float*)d_in[1];
  const float* b_qkv  = (const float*)d_in[2];
  const float* q_w    = (const float*)d_in[3];
  const float* k_w    = (const float*)d_in[4];
  const float* w_proj = (const float*)d_in[5];
  const float* b_proj = (const float*)d_in[6];
  float* out = (float*)d_out;

  char* ws = (char*)d_ws;
  unsigned short* xbf     = (unsigned short*)(ws);                  //  8.0 MB [B*N,C]
  unsigned short* wqkvT   = (unsigned short*)(ws +  8388608);       //  6.0 MB [3C,C]
  unsigned short* wprojT  = (unsigned short*)(ws + 14680064);       //  2.0 MB [C,C]
  unsigned short* qbf     = (unsigned short*)(ws + 16777216);       //  8.0 MB [BH,N,D]
  unsigned short* kbf     = (unsigned short*)(ws + 25165824);       //  8.0 MB [BH,N,D]
  unsigned short* vtbf    = (unsigned short*)(ws + 33554432);       //  8.0 MB [BH,D,N]
  unsigned short* attnbf  = (unsigned short*)(ws + 41943040);       //  8.0 MB [B*N,C]

  cvt_kernel<<<(B_ * N_ * C_) / 4 / 256, 256, 0, stream>>>(x, xbf, B_ * N_ * C_);
  cvt_transpose_kernel<<<dim3(C_ / 32, 3 * C_ / 32), 256, 0, stream>>>(w_qkv, wqkvT, C_, 3 * C_);
  cvt_transpose_kernel<<<dim3(C_ / 32, C_ / 32), 256, 0, stream>>>(w_proj, wprojT, C_, C_);

  qkv_gemm_kernel<<<dim3(32, 48), 256, 0, stream>>>(xbf, wqkvT, b_qkv, q_w, k_w,
                                                    qbf, kbf, vtbf);
  flash_kernel<<<dim3(16, 32), 256, 0, stream>>>(qbf, kbf, vtbf, attnbf);
  proj_gemm_kernel<<<dim3(32, 16), 256, 0, stream>>>(attnbf, wprojT, b_proj, out);
}
